// TTCN_22222160789763
// MI455X (gfx1250) — matrix operand order repaired and checked by reading, not compiled, not run on hardware
//
#include <hip/hip_runtime.h>
#include <hip/hip_bf16.h>

// Fused TTCN head for MI455X (gfx1250, wave32):
//   per (b,o): h = relu(z[b] @ W1[o]^T + b1[o]); logits = h @ W2[o]^T + b2[o];
//   masked column softmax over L; out[b,o] = sum_{l,m} w[l,m] * z[b,l,m].
//
// Compute-bound (137 GFLOP vs 71 MB unique bytes) -> bf16 WMMA 16x16x32 / f32 acc.
// Fast path (ws big enough): pre-pass converts W1/W2 to bf16 in WMMA-B tile layout
// and z to bf16 in d_ws; main kernel stages tiles with the Tensor Data Mover
// (tensor_load_to_lds, double-buffered, s_wait_tensorcnt) so DMA overlaps WMMA.
// Inner loop interleaves two row-tiles per wave: consecutive WMMAs hit
// independent accumulators (no RAW-hazard NOPs) and each B tile is read once.
// Fallback: per-block VALU conversion staging.

#define B_    16
#define L_    256
#define DIN_  256
#define DOUT_ 128
#define NEGV  (-1e9f)

// ws layout (bf16 elements): [0) W1 tiles, [8388608) W2 tiles, [16777216) z
#define WS_W2_OFF   8388608u
#define WS_Z_OFF    16777216u
#define WS_NEED_B   35651584ull   // 33.55MB weights + 2MB z

typedef __attribute__((ext_vector_type(16))) __bf16 v16bf;
typedef __attribute__((ext_vector_type(8)))  float  v8f;
typedef __attribute__((ext_vector_type(4)))  unsigned int u32x4;
typedef __attribute__((ext_vector_type(8)))  int    i32x8;
typedef __attribute__((ext_vector_type(4)))  int    i32x4;

union ABu { v16bf v; uint4 q[2]; };

// ---- TDM: DMA a d0 x d1 tile of 2-byte elements (tensor==tile, row stride d0)
// from global bf16 array into LDS. D# bitfields per CDNA5 ISA ch.8.
// 6-arg builtin form (clang-23 / therock headers): (g0, g1, g2, g3, g4, cpol).
__device__ __forceinline__ void tdm_load(const __bf16* gsrc, void* ldst,
                                         unsigned int d0, unsigned int d1) {
    unsigned long long ga = (unsigned long long)gsrc;   // 57-bit global byte addr
    unsigned int ldsOff   = (unsigned int)(unsigned long long)ldst; // LDS aperture: addr[31:0]
    u32x4 g0;
    g0.x = 1u;                                            // count=1, user D#, no gather
    g0.y = ldsOff;                                        // lds_addr
    g0.z = (unsigned int)(ga & 0xFFFFFFFFu);              // global_addr[31:0]
    g0.w = (unsigned int)((ga >> 32) & 0x01FFFFFFu) | (2u << 30); // addr[56:32] | type=2
    i32x8 g1;
    g1[0] = 0x00010000;                                   // data_size=1 (2B), mask=0
    g1[1] = (int)((d0 & 0xFFFFu) << 16);                  // tensor_dim0[15:0]
    g1[2] = (int)((d0 >> 16) | ((d1 & 0xFFFFu) << 16));   // dim0 hi | tensor_dim1 lo
    g1[3] = (int)((d1 >> 16) | ((d0 & 0xFFFFu) << 16));   // dim1 hi | tile_dim0
    g1[4] = (int)(d1 & 0xFFFFu);                          // tile_dim1 (0/1 => 1D)
    g1[5] = (int)d0;                                      // tensor_dim0_stride lo
    g1[6] = 0;                                            // stride hi | dim1_stride lo
    g1[7] = 0;
    i32x4 zz = {0, 0, 0, 0};                              // groups 2/3 unused (<=2D)
    i32x8 z8 = {0, 0, 0, 0, 0, 0, 0, 0};
    __builtin_amdgcn_tensor_load_to_lds(g0, g1, zz, zz, z8, 0);
}

// A operand (16x32 bf16 from LDS matrix, row stride 256): lane<16 holds row
// M=row0+lane, K {ib+0..7, ib+16..23}; lane>=16 same row, K {+8..15, +24..31}.
__device__ __forceinline__ v16bf load_a_lds(const __bf16* M, int row0, int ib, int lane) {
    int lr = lane & 15, hi = lane >> 4;
    const __bf16* p = M + (row0 + lr) * 256 + ib + hi * 8;
    ABu u;
    u.q[0] = *(const uint4*)(p);
    u.q[1] = *(const uint4*)(p + 16);
    return u.v;
}

// B operand (32x16 bf16): lane column N=lane&15, K {ib+8h+0..7, ib+16+8h+0..7}
// (h=lane>>4). Tile stored transposed as wt[i][n] (row = 32B) -> strided gather.
__device__ __forceinline__ v16bf load_b_lds(const __bf16* Wt, int ib, int lane) {
    const __bf16* p = Wt + (ib + (lane >> 4) * 8) * 16 + (lane & 15);
    v16bf v;
    #pragma unroll
    for (int s = 0; s < 8; ++s) {
        v[s]     = p[s * 16];
        v[s + 8] = p[(s + 16) * 16];
    }
    return v;
}

// ---- Pre-pass: fp32 -> bf16, weights into transposed contiguous 8KB tiles ----
extern "C" __global__ void __launch_bounds__(256)
ttcn_convert_kernel(const float* __restrict__ z, const float* __restrict__ W1,
                    const float* __restrict__ W2, __bf16* __restrict__ wbf)
{
    const int tid = threadIdx.x;
    const int blk = blockIdx.x;
    if (blk < 4096) {                       // weight tiles: 2 stages x 128 o x 16 kt
        int stage = blk >> 11;
        int rem   = blk & 2047;             // o*16 + kt
        const float* src = (stage ? W2 : W1) + (size_t)rem * 4096;
        __bf16* dst = wbf + (size_t)stage * WS_W2_OFF + (size_t)rem * 4096;
        #pragma unroll 4
        for (int r = 0; r < 16; ++r) {
            int idx = r * 256 + tid;        // coalesced fp32 read
            dst[(idx & 255) * 16 + (idx >> 8)] = (__bf16)src[idx]; // [i][k] tile
        }
    } else {                                // z: 256 blocks x 4096 elements, linear
        int zb = blk - 4096;
        const float* src = z + (size_t)zb * 4096;
        __bf16* dst = wbf + WS_Z_OFF + (size_t)zb * 4096;
        #pragma unroll 4
        for (int r = 0; r < 16; ++r) {
            int idx = r * 256 + tid;
            dst[idx] = (__bf16)src[idx];
        }
    }
}

extern "C" __global__ void __launch_bounds__(256, 1)
ttcn_fused_kernel(const float* __restrict__ z, const unsigned char* __restrict__ mask,
                  const float* __restrict__ W1, const float* __restrict__ b1,
                  const float* __restrict__ W2, const float* __restrict__ b2,
                  float* __restrict__ out, const __bf16* __restrict__ wbf, int useTdm)
{
    extern __shared__ char smem[];
    __bf16* zbf = (__bf16*)(smem);            // 256x256 bf16  (128 KB)
    __bf16* hbf = (__bf16*)(smem + 131072);   // 256x256 bf16  (128 KB)
    __bf16* wtb = (__bf16*)(smem + 262144);   // 2 x (256x16) bf16 tile buffers (16 KB)
    float*  lg  = (float* )(smem + 278528);   // 256x17 f32 logits slab (pad=1)
    float*  acs = (float* )(smem + 295936);   // scalar accumulator

    const int tid  = threadIdx.x;
    const int lane = tid & 31;
    const int wv   = tid >> 5;
    const int b    = blockIdx.x & 15;   // b in low bits: 16 consecutive WGs share W[o] in L2
    const int o    = blockIdx.x >> 4;

    if (tid == 0) *acs = 0.0f;

    // ---- Stage z[b] into LDS as bf16 ----
    if (useTdm) {
        if (wv == 0) {                       // one wave drives the TDM (EXEC-independent DMA)
            tdm_load(wbf + WS_Z_OFF + (size_t)b * 65536, zbf, 256, 256);
            __builtin_amdgcn_s_wait_tensorcnt(0);
        }
    } else {
        const float4* z4 = (const float4*)(z + (size_t)b * (L_ * DIN_));
        for (int r = 0; r < 64; ++r) {
            int idx = r * 256 + tid;
            float4 v = z4[idx];
            __bf16 t4[4] = {(__bf16)v.x, (__bf16)v.y, (__bf16)v.z, (__bf16)v.w};
            *(uint2*)(zbf + idx * 4) = *(const uint2*)t4;
        }
    }
    __syncthreads();

    const float*  w1b = W1 + (size_t)o * (DIN_ * DIN_);
    const float*  w2b = W2 + (size_t)o * (DIN_ * DIN_);
    const __bf16* wq1 = wbf + (size_t)o * 16 * 4096;            // pre-tiled bf16 W1[o]
    const __bf16* wq2 = wbf + WS_W2_OFF + (size_t)o * 16 * 4096;
    const int row0 = wv * 32;                // each wave owns 32 output rows (2 tiles)
    const int n  = lane & 15;
    const int hi = lane >> 4;

    // ======== GEMM1: h = relu(z @ W1^T + b1), bf16 result to LDS ========
    if (useTdm && wv == 0) tdm_load(wq1, wtb, 4096, 1);   // prologue: tile 0 -> buf0
    for (int kt = 0; kt < 16; ++kt) {
        __bf16* cur = wtb + (kt & 1) * 4096;
        if (useTdm) {
            if (wv == 0) {                   // issue kt+1, then wait for kt (<=1 in flight)
                if (kt + 1 < 16) {
                    tdm_load(wq1 + (kt + 1) * 4096, wtb + ((kt + 1) & 1) * 4096, 4096, 1);
                    __builtin_amdgcn_s_wait_tensorcnt(1);
                } else {
                    __builtin_amdgcn_s_wait_tensorcnt(0);
                }
            }
        } else {
            const float* src = w1b + kt * 4096;
            #pragma unroll 4
            for (int r = 0; r < 16; ++r) {
                int idx = r * 256 + tid;
                cur[(idx & 255) * 16 + (idx >> 8)] = (__bf16)src[idx];
            }
            if (kt + 1 < 16) __builtin_prefetch(w1b + (kt + 1) * 4096 + tid * 16, 0, 0);
        }
        __syncthreads();

        float bias = b1[o * 256 + kt * 16 + n];
        v8f acc0 = {}, acc1 = {};
        #pragma unroll
        for (int ib = 0; ib < 8; ++ib) {     // interleaved dual-tile chain: B read once,
            v16bf a0 = load_a_lds(zbf, row0,      ib * 32, lane);
            v16bf a1 = load_a_lds(zbf, row0 + 16, ib * 32, lane);
            v16bf bv = load_b_lds(cur, ib * 32, lane);
            acc0 = __builtin_amdgcn_wmma_f32_16x16x32_bf16(
                       false, a0, false, bv, (short)0, acc0, false, false);
            acc1 = __builtin_amdgcn_wmma_f32_16x16x32_bf16(
                       false, a1, false, bv, (short)0, acc1, false, false);
        }
        #pragma unroll
        for (int r = 0; r < 8; ++r) {        // C/D: lane<16 -> M=r else M=r+8; N=lane&15
            float v0 = acc0[r] + bias;
            float v1 = acc1[r] + bias;
            v0 = v0 > 0.0f ? v0 : 0.0f;
            v1 = v1 > 0.0f ? v1 : 0.0f;
            hbf[(row0 +      r + hi * 8) * 256 + kt * 16 + n] = (__bf16)v0;
            hbf[(row0 + 16 + r + hi * 8) * 256 + kt * 16 + n] = (__bf16)v1;
        }
        __syncthreads();
    }

    // ======== GEMM2 + masked softmax over L + weighted z reduction ========
    const int col = tid >> 4;                // m column within tile (0..15)
    const int s   = tid & 15;                // reducer lane within 16-lane group

    if (useTdm && wv == 0) tdm_load(wq2, wtb, 4096, 1);
    for (int mt = 0; mt < 16; ++mt) {
        __bf16* cur = wtb + (mt & 1) * 4096;
        if (useTdm) {
            if (wv == 0) {
                if (mt + 1 < 16) {
                    tdm_load(wq2 + (mt + 1) * 4096, wtb + ((mt + 1) & 1) * 4096, 4096, 1);
                    __builtin_amdgcn_s_wait_tensorcnt(1);
                } else {
                    __builtin_amdgcn_s_wait_tensorcnt(0);
                }
            }
        } else {
            const float* src = w2b + mt * 4096;
            #pragma unroll 4
            for (int r = 0; r < 16; ++r) {
                int idx = r * 256 + tid;
                cur[(idx & 255) * 16 + (idx >> 8)] = (__bf16)src[idx];
            }
            if (mt + 1 < 16) __builtin_prefetch(w2b + (mt + 1) * 4096 + tid * 16, 0, 0);
        }
        __syncthreads();

        float bb = b2[o * 256 + mt * 16 + n];
        v8f acc0 = {}, acc1 = {};
        #pragma unroll
        for (int kb = 0; kb < 8; ++kb) {
            v16bf a0 = load_a_lds(hbf, row0,      kb * 32, lane);
            v16bf a1 = load_a_lds(hbf, row0 + 16, kb * 32, lane);
            v16bf bv = load_b_lds(cur, kb * 32, lane);
            acc0 = __builtin_amdgcn_wmma_f32_16x16x32_bf16(
                       false, a0, false, bv, (short)0, acc0, false, false);
            acc1 = __builtin_amdgcn_wmma_f32_16x16x32_bf16(
                       false, a1, false, bv, (short)0, acc1, false, false);
        }
        #pragma unroll
        for (int r = 0; r < 8; ++r) {
            int l0 = row0 +      r + hi * 8;
            int l1 = row0 + 16 + r + hi * 8;
            float v0 = acc0[r] + bb;
            float v1 = acc1[r] + bb;
            if (!mask[b * 256 + l0]) v0 = NEGV;
            if (!mask[b * 256 + l1]) v1 = NEGV;
            lg[l0 * 17 + n] = v0;            // pad=1 -> conflict-free column reads
            lg[l1 * 17 + n] = v1;
        }
        __syncthreads();

        // Column softmax over l (256 rows), fused with sum_l w*z. 16 threads/col.
        float mx = -3.0e38f;
        #pragma unroll
        for (int j = 0; j < 16; ++j)
            mx = fmaxf(mx, lg[(s + 16 * j) * 17 + col]);
        #pragma unroll
        for (int off = 1; off < 16; off <<= 1)
            mx = fmaxf(mx, __shfl_xor(mx, off, 16));

        float se = 0.0f, wz = 0.0f;
        #pragma unroll
        for (int j = 0; j < 16; ++j) {
            int l = s + 16 * j;
            float e = __expf(lg[l * 17 + col] - mx);
            se += e;
            wz += e * (float)zbf[l * 256 + mt * 16 + col];
        }
        #pragma unroll
        for (int off = 1; off < 16; off <<= 1) {
            se += __shfl_xor(se, off, 16);
            wz += __shfl_xor(wz, off, 16);
        }
        if (s == 0) atomicAdd(acs, wz / se);
        __syncthreads();
    }

    if (tid == 0) out[b * DOUT_ + o] = *acs;
}

extern "C" void kernel_launch(void* const* d_in, const int* in_sizes, int n_in,
                              void* d_out, int out_size, void* d_ws, size_t ws_size,
                              hipStream_t stream) {
    const float*         z    = (const float*)d_in[0];
    const unsigned char* mask = (const unsigned char*)d_in[1];
    const float*         W1   = (const float*)d_in[2];
    const float*         b1   = (const float*)d_in[3];
    const float*         W2   = (const float*)d_in[4];
    const float*         b2   = (const float*)d_in[5];
    float*               out  = (float*)d_out;

    const int useTdm = (ws_size >= WS_NEED_B) ? 1 : 0;   // constant across calls
    __bf16* wbf = (__bf16*)d_ws;

    if (useTdm) {
        // One cheap sweep: 71MB read / 35.7MB write @ 23.3TB/s ~ a few us.
        ttcn_convert_kernel<<<dim3(4096 + 256), dim3(256), 0, stream>>>(z, W1, W2, wbf);
    }

    const size_t ldsBytes = 296000;  // zbf + hbf + 2 tile bufs + logits + acc
    ttcn_fused_kernel<<<dim3(B_ * DOUT_), dim3(256), ldsBytes, stream>>>(
        z, mask, W1, b1, W2, b2, out, wbf, useTdm);
}
